// AttentionBlock_30717606101281
// MI455X (gfx1250) — compile-verified, matrix operand order checked by reading, not run on hardware
//
#include <hip/hip_runtime.h>
#include <hip/hip_bf16.h>

// ---------------------------------------------------------------------------
// Fused transformer block for MI455X (gfx1250, wave32, WMMA).
// One workgroup (256 threads = 8 waves) processes one batch element b:
//   async x->LDS copy -> LN1 -> QKV (wmma) -> causal attention (wmma + shfl
//   softmax) -> O-proj (wmma) + residual -> LN2 -> FFN 128->512->128 (wmma)
//   + residual -> async LDS->out copy.
// Everything stays in LDS; only x is read and out written to HBM (coalesced,
// via the gfx1250 async global<->LDS path when available).
// ---------------------------------------------------------------------------

typedef __attribute__((ext_vector_type(16))) _Float16 v16h;
typedef __attribute__((ext_vector_type(8)))  float    v8f;
typedef int v4i __attribute__((vector_size(16)));  // matches builtin V4i

#define AS1 __attribute__((address_space(1)))
#define AS3 __attribute__((address_space(3)))

#if __has_builtin(__builtin_amdgcn_global_load_async_to_lds_b128) && \
    __has_builtin(__builtin_amdgcn_global_store_async_from_lds_b128) && \
    __has_builtin(__builtin_amdgcn_s_wait_asynccnt)
#define HAS_ASYNC_LDS 1
#else
#define HAS_ASYNC_LDS 0
#endif

namespace {

constexpr int Tn = 128;   // sequence length
constexpr int Dn = 128;   // model dim
constexpr int HSn = 16;   // head size
constexpr int DF = 512;   // ffn dim

// LDS layout (dynamic shared memory)
constexpr int XRES_OFF = 0;                       // 128x128 f32      = 65536 B
constexpr int H_OFF    = XRES_OFF + 128 * 128 * 4;// 128x136 f16      = 34816 B
constexpr int O_OFF    = H_OFF + 128 * 136 * 2;   // 128x136 f16      = 34816 B
constexpr int POOL_OFF = O_OFF + 128 * 136 * 2;   // per-wave scratch / ffn act
constexpr int CHUNK    = 16640;                   // per-wave: Q 4096 | Kt 4224 | V 4096 | wei 4224
constexpr int POOL_BYTES = 128 * 520 * 2;         // = 133120 = 8*CHUNK (aliased)
constexpr int SMEM_BYTES = POOL_OFF + POOL_BYTES; // 268288 B  (< 320 KB WGP LDS)

// K-index of element e of a 16-wide f16 fragment for this lane (ISA 7.12.2,
// 16-bit A-matrix 16x32 layout; B 32x16 uses the mirrored mapping).
__device__ __forceinline__ int kof(int e, int lane) {
  return (e & 7) | ((lane & 16) >> 1) | ((e & 8) << 1);
}
// Row of C/D element i for this lane (32-bit C/D 16x16 layout).
__device__ __forceinline__ int rof(int i, int lane) {
  return i + ((lane & 16) >> 1);
}

__device__ __forceinline__ v8f wmma_f16(v16h a, v16h b, v8f c) {
  return __builtin_amdgcn_wmma_f32_16x16x32_f16(
      /*neg_a=*/false, a, /*neg_b=*/false, b,
      /*c_mod=*/(short)0, c, /*reuse_a=*/false, /*reuse_b=*/false);
}

// A fragment (16x32 f16) from row-major f16 LDS tile at (m0, k0).
__device__ __forceinline__ v16h lds_load_a(const _Float16* base, int stride,
                                           int m0, int k0, int lane) {
  v16h a;
  const int r = m0 + (lane & 15);
#pragma unroll
  for (int e = 0; e < 16; ++e) a[e] = base[r * stride + k0 + kof(e, lane)];
  return a;
}

// A fragment with K zero-padding (k >= klimit -> 0).
__device__ __forceinline__ v16h lds_load_a_masked(const _Float16* base, int stride,
                                                  int m0, int k0, int klimit,
                                                  int lane) {
  v16h a;
  const int r = m0 + (lane & 15);
#pragma unroll
  for (int e = 0; e < 16; ++e) {
    const int k = k0 + kof(e, lane);
    a[e] = (k < klimit) ? base[r * stride + k] : (_Float16)0.f;
  }
  return a;
}

// B fragment (32x16 f16) from row-major f16 LDS matrix [k][n] at (k0, n0).
__device__ __forceinline__ v16h lds_load_b(const _Float16* base, int stride,
                                           int k0, int n0, int lane) {
  v16h b;
  const int c = n0 + (lane & 15);
#pragma unroll
  for (int e = 0; e < 16; ++e) b[e] = base[(k0 + kof(e, lane)) * stride + c];
  return b;
}

__device__ __forceinline__ v16h lds_load_b_masked(const _Float16* base, int stride,
                                                  int k0, int n0, int klimit,
                                                  int lane) {
  v16h b;
  const int c = n0 + (lane & 15);
#pragma unroll
  for (int e = 0; e < 16; ++e) {
    const int k = k0 + kof(e, lane);
    b[e] = (k < klimit) ? base[k * stride + c] : (_Float16)0.f;
  }
  return b;
}

// B fragment converted on the fly from a row-major f32 global matrix [k][n].
__device__ __forceinline__ v16h gmem_load_b_f32(const float* W, int ldw,
                                                int k0, int n0, int lane) {
  v16h b;
  const int c = n0 + (lane & 15);
#pragma unroll
  for (int e = 0; e < 16; ++e)
    b[e] = (_Float16)W[(size_t)(k0 + kof(e, lane)) * ldw + c];
  return b;
}

}  // namespace

__global__ __launch_bounds__(256, 1) void attn_block_kernel(
    const float* __restrict__ x,  const float* __restrict__ Wq,
    const float* __restrict__ Wk, const float* __restrict__ Wv,
    const float* __restrict__ Wo, const float* __restrict__ bo,
    const float* __restrict__ W1, const float* __restrict__ b1,
    const float* __restrict__ W2, const float* __restrict__ b2,
    const float* __restrict__ g1, const float* __restrict__ be1,
    const float* __restrict__ g2, const float* __restrict__ be2,
    float* __restrict__ out) {
  extern __shared__ char smem[];
  float*    xres = (float*)(smem + XRES_OFF);     // residual stream, f32
  _Float16* hbuf = (_Float16*)(smem + H_OFF);     // LN output, f16, stride 136
  _Float16* obuf = (_Float16*)(smem + O_OFF);     // attn out,  f16, stride 136
  char*     pool = smem + POOL_OFF;

  const int tid  = threadIdx.x;
  const int lane = tid & 31;
  const int wv   = tid >> 5;     // wave id 0..7  (== head id in attention)
  const int b    = blockIdx.x;
  const float* xb = x + (size_t)b * Tn * Dn;

  if (tid == 0) {  // warm L2 path for the big shared weights
    __builtin_prefetch(W1, 0, 0);
    __builtin_prefetch(W2, 0, 0);
  }

  // ---------------- Phase 0: coalesced x[b] -> LDS (async, 16 B/lane) -------
  {
    const float4* xb4 = (const float4*)xb;
    float4* xr4 = (float4*)xres;
#if HAS_ASYNC_LDS
#pragma unroll
    for (int it = 0; it < 16; ++it) {
      const int idx = it * 256 + tid;  // 4096 float4 chunks total
      __builtin_amdgcn_global_load_async_to_lds_b128(
          (AS1 v4i*)(xb4 + idx), (AS3 v4i*)(xr4 + idx), 0, 0);
    }
    __builtin_amdgcn_s_wait_asynccnt(0);
#else
#pragma unroll
    for (int it = 0; it < 16; ++it) {
      const int idx = it * 256 + tid;
      xr4[idx] = xb4[idx];
    }
#endif
  }
  __syncthreads();

  // ---------------- Phase 1: LayerNorm 1 -> hbuf (f16), 2 lanes per row -----
  {
    const int row = tid >> 1;
    const int c0  = (tid & 1) * 64;
    float s = 0.f, ss = 0.f;
#pragma unroll 8
    for (int c = 0; c < 64; ++c) {
      float v = xres[row * 128 + c0 + c];
      s += v; ss += v * v;
    }
    s  += __shfl_xor(s, 1, 32);   // lane pairs (tid, tid^1) share one row
    ss += __shfl_xor(ss, 1, 32);
    const float mu = s * (1.f / Dn);
    const float var = ss * (1.f / Dn) - mu * mu;
    const float rstd = rsqrtf(var + 1e-5f);
#pragma unroll 8
    for (int c = 0; c < 64; ++c) {
      float v = (xres[row * 128 + c0 + c] - mu) * rstd * g1[c0 + c] + be1[c0 + c];
      hbuf[row * 136 + c0 + c] = (_Float16)v;
    }
  }
  __syncthreads();

  // ---------------- Phase 2: attention, one head per wave -------------------
  {
    _Float16* Qst  = (_Float16*)(pool + wv * CHUNK);           // [t][e]  ld 16
    _Float16* Kt   = (_Float16*)(pool + wv * CHUNK + 4096);    // [e][s]  ld 132
    _Float16* Vst  = (_Float16*)(pool + wv * CHUNK + 8320);    // [s][e]  ld 16
    _Float16* weiS = (_Float16*)(pool + wv * CHUNK + 12416);   // [t][s]  ld 132

    const float* wq = Wq + (size_t)wv * Dn * HSn;  // (D, HS) row-major
    const float* wk = Wk + (size_t)wv * Dn * HSn;
    const float* wvv = Wv + (size_t)wv * Dn * HSn;

    v16h Bq[4], Bk[4], Bv[4];
#pragma unroll
    for (int ks = 0; ks < 4; ++ks) {
      Bq[ks] = gmem_load_b_f32(wq,  HSn, ks * 32, 0, lane);
      Bk[ks] = gmem_load_b_f32(wk,  HSn, ks * 32, 0, lane);
      Bv[ks] = gmem_load_b_f32(wvv, HSn, ks * 32, 0, lane);
    }

    // 2a: Q,K,V = h @ W  (K staged transposed for the score GEMM)
    for (int mt = 0; mt < 8; ++mt) {
      v8f aq = {}, ak = {}, av = {};
#pragma unroll
      for (int ks = 0; ks < 4; ++ks) {
        v16h A = lds_load_a(hbuf, 136, mt * 16, ks * 32, lane);
        aq = wmma_f16(A, Bq[ks], aq);
        ak = wmma_f16(A, Bk[ks], ak);
        av = wmma_f16(A, Bv[ks], av);
      }
#pragma unroll
      for (int i = 0; i < 8; ++i) {
        const int r = mt * 16 + rof(i, lane);
        const int c = lane & 15;
        Qst[r * 16 + c]  = (_Float16)aq[i];
        Kt[c * 132 + r]  = (_Float16)ak[i];
        Vst[r * 16 + c]  = (_Float16)av[i];
      }
    }

    // 2b: causal scores + softmax + (wei @ V), streamed over 16-row bands
    for (int mt = 0; mt < 8; ++mt) {
      v16h Qa = lds_load_a_masked(Qst, 16, mt * 16, 0, HSn, lane);  // K pad 16->32
      v8f sc[8];
#pragma unroll
      for (int st = 0; st < 8; ++st) {
        if (st <= mt) {
          v16h Bf = lds_load_b_masked(Kt, 132, 0, st * 16, HSn, lane);
          v8f z = {};
          sc[st] = wmma_f16(Qa, Bf, z);
        }
      }
      // scale + causal mask
#pragma unroll
      for (int st = 0; st < 8; ++st) {
        if (st <= mt) {
#pragma unroll
          for (int i = 0; i < 8; ++i) {
            const int t = mt * 16 + rof(i, lane);
            const int s = st * 16 + (lane & 15);
            const float v = sc[st][i] * 0.25f;  // 16^-0.5
            sc[st][i] = (s <= t) ? v : -1e30f;
          }
        }
      }
      // softmax across the 128-wide row (16 lanes x up-to-8 fragments)
      float mx[8], sm[8];
#pragma unroll
      for (int i = 0; i < 8; ++i) mx[i] = -1e30f;
#pragma unroll
      for (int st = 0; st < 8; ++st)
        if (st <= mt)
#pragma unroll
          for (int i = 0; i < 8; ++i) mx[i] = fmaxf(mx[i], sc[st][i]);
#pragma unroll
      for (int i = 0; i < 8; ++i) {
#pragma unroll
        for (int d = 1; d < 16; d <<= 1)
          mx[i] = fmaxf(mx[i], __shfl_xor(mx[i], d, 32));
        sm[i] = 0.f;
      }
#pragma unroll
      for (int st = 0; st < 8; ++st)
        if (st <= mt)
#pragma unroll
          for (int i = 0; i < 8; ++i) {
            const float e = __expf(sc[st][i] - mx[i]);
            sc[st][i] = e;
            sm[i] += e;
          }
#pragma unroll
      for (int i = 0; i < 8; ++i) {
#pragma unroll
        for (int d = 1; d < 16; d <<= 1) sm[i] += __shfl_xor(sm[i], d, 32);
        sm[i] = 1.f / sm[i];
      }
      // stage wei (f16) for layout conversion C -> A
#pragma unroll
      for (int st = 0; st < 8; ++st)
        if (st <= mt)
#pragma unroll
          for (int i = 0; i < 8; ++i)
            weiS[rof(i, lane) * 132 + st * 16 + (lane & 15)] =
                (_Float16)(sc[st][i] * sm[i]);

      // O-band = wei @ V  (K truncated to the causal extent)
      v8f ao = {};
      const int klim = (mt + 1) * 16;
#pragma unroll
      for (int ks = 0; ks < 4; ++ks) {
        if (ks * 32 < klim) {
          v16h Aw = lds_load_a_masked(weiS, 132, 0, ks * 32, klim, lane);
          v16h Bf = lds_load_b(Vst, 16, ks * 32, 0, lane);
          ao = wmma_f16(Aw, Bf, ao);
        }
      }
#pragma unroll
      for (int i = 0; i < 8; ++i)
        obuf[(mt * 16 + rof(i, lane)) * 136 + wv * 16 + (lane & 15)] =
            (_Float16)ao[i];
    }
  }
  __syncthreads();

  // ---------------- Phase 3: x += O @ Wo + bo  (wave wv -> col tile wv) -----
  {
    const int n0 = wv * 16;
    v16h Bo[4];
#pragma unroll
    for (int ks = 0; ks < 4; ++ks)
      Bo[ks] = gmem_load_b_f32(Wo, Dn, ks * 32, n0, lane);
    const float bov = bo[n0 + (lane & 15)];
    for (int mt = 0; mt < 8; ++mt) {
      v8f acc = {};
#pragma unroll
      for (int ks = 0; ks < 4; ++ks)
        acc = wmma_f16(lds_load_a(obuf, 136, mt * 16, ks * 32, lane), Bo[ks], acc);
#pragma unroll
      for (int i = 0; i < 8; ++i) {
        const int r = mt * 16 + rof(i, lane);
        const int c = n0 + (lane & 15);
        xres[r * 128 + c] += acc[i] + bov;
      }
    }
  }
  __syncthreads();

  // ---------------- Phase 4: LayerNorm 2 -> hbuf, 2 lanes per row -----------
  {
    const int row = tid >> 1;
    const int c0  = (tid & 1) * 64;
    float s = 0.f, ss = 0.f;
#pragma unroll 8
    for (int c = 0; c < 64; ++c) {
      float v = xres[row * 128 + c0 + c];
      s += v; ss += v * v;
    }
    s  += __shfl_xor(s, 1, 32);
    ss += __shfl_xor(ss, 1, 32);
    const float mu = s * (1.f / Dn);
    const float var = ss * (1.f / Dn) - mu * mu;
    const float rstd = rsqrtf(var + 1e-5f);
#pragma unroll 8
    for (int c = 0; c < 64; ++c) {
      float v = (xres[row * 128 + c0 + c] - mu) * rstd * g2[c0 + c] + be2[c0 + c];
      hbuf[row * 136 + c0 + c] = (_Float16)v;
    }
  }
  __syncthreads();

  // ---------------- Phase 5: g = relu(h2 @ W1 + b1)  (128x512, f16) ---------
  {
    _Float16* g = (_Float16*)pool;  // stride 520 (aliases attention scratch)
#pragma unroll
    for (int j = 0; j < 4; ++j) {
      const int n0 = (wv + j * 8) * 16;
      v16h B1f[4];
#pragma unroll
      for (int ks = 0; ks < 4; ++ks)
        B1f[ks] = gmem_load_b_f32(W1, DF, ks * 32, n0, lane);
      const float b1v = b1[n0 + (lane & 15)];
      for (int mt = 0; mt < 8; ++mt) {
        v8f acc = {};
#pragma unroll
        for (int ks = 0; ks < 4; ++ks)
          acc = wmma_f16(lds_load_a(hbuf, 136, mt * 16, ks * 32, lane), B1f[ks], acc);
#pragma unroll
        for (int i = 0; i < 8; ++i) {
          float v = acc[i] + b1v;
          v = v > 0.f ? v : 0.f;  // relu
          g[(mt * 16 + rof(i, lane)) * 520 + n0 + (lane & 15)] = (_Float16)v;
        }
      }
    }
  }
  __syncthreads();

  // ---------------- Phase 6: xres = x + g @ W2 + b2 (in place) --------------
  {
    const _Float16* g = (const _Float16*)pool;
    const int n0 = wv * 16;
    v16h B2f[16];
#pragma unroll
    for (int ks = 0; ks < 16; ++ks)
      B2f[ks] = gmem_load_b_f32(W2, Dn, ks * 32, n0, lane);
    const float b2v = b2[n0 + (lane & 15)];
    for (int mt = 0; mt < 8; ++mt) {
      v8f acc = {};
#pragma unroll
      for (int ks = 0; ks < 16; ++ks)
        acc = wmma_f16(lds_load_a(g, 520, mt * 16, ks * 32, lane), B2f[ks], acc);
#pragma unroll
      for (int i = 0; i < 8; ++i) {
        const int r = mt * 16 + rof(i, lane);
        const int c = n0 + (lane & 15);
        xres[r * 128 + c] = xres[r * 128 + c] + acc[i] + b2v;  // each elem owned
      }
    }
  }
  __syncthreads();

  // ---------------- Phase 7: coalesced LDS -> out[b] (async, 16 B/lane) -----
  {
    float4* outp4 = (float4*)(out + (size_t)b * Tn * Dn);
    const float4* xr4 = (const float4*)xres;
#if HAS_ASYNC_LDS
#pragma unroll
    for (int it = 0; it < 16; ++it) {
      const int idx = it * 256 + tid;
      __builtin_amdgcn_global_store_async_from_lds_b128(
          (AS1 v4i*)(outp4 + idx), (AS3 v4i*)(xr4 + idx), 0, 0);
    }
    __builtin_amdgcn_s_wait_asynccnt(0);  // (S_ENDPGM also implies wait-idle)
#else
#pragma unroll
    for (int it = 0; it < 16; ++it) {
      const int idx = it * 256 + tid;
      outp4[idx] = xr4[idx];
    }
#endif
  }
}

extern "C" void kernel_launch(void* const* d_in, const int* in_sizes, int n_in,
                              void* d_out, int out_size, void* d_ws, size_t ws_size,
                              hipStream_t stream) {
  (void)n_in; (void)out_size; (void)d_ws; (void)ws_size;
  const float* x   = (const float*)d_in[0];
  const float* Wq  = (const float*)d_in[1];
  const float* Wk  = (const float*)d_in[2];
  const float* Wv  = (const float*)d_in[3];
  const float* Wo  = (const float*)d_in[4];
  const float* bo  = (const float*)d_in[5];
  const float* W1  = (const float*)d_in[6];
  const float* b1  = (const float*)d_in[7];
  const float* W2  = (const float*)d_in[8];
  const float* b2  = (const float*)d_in[9];
  const float* g1  = (const float*)d_in[10];
  const float* be1 = (const float*)d_in[11];
  const float* g2  = (const float*)d_in[12];
  const float* be2 = (const float*)d_in[13];
  float* outp = (float*)d_out;

  const int Bn = in_sizes[0] / (Tn * Dn);  // 1024

  // Allow large dynamic LDS (WGP has 320 KB); harmless if it no-ops.
  (void)hipFuncSetAttribute((const void*)attn_block_kernel,
                            hipFuncAttributeMaxDynamicSharedMemorySize,
                            SMEM_BYTES);

  attn_block_kernel<<<dim3(Bn), dim3(256), SMEM_BYTES, stream>>>(
      x, Wq, Wk, Wv, Wo, bo, W1, b1, W2, b2, g1, be1, g2, be2, outp);
}
